// BahdanauAttention_85143431676471
// MI455X (gfx1250) — matrix operand order repaired and checked by reading, not compiled, not run on hardware
//
#include <hip/hip_runtime.h>
#include <math.h>

#define Bb 64
#define Tt 2048
#define Ee 512
#define Dd 1024
#define Hh 128
#define LDA 80   // LDS row stride in halfwords: rows 16B-aligned, banks spread

typedef __attribute__((ext_vector_type(16))) __bf16       v16bf;
typedef __attribute__((ext_vector_type(8)))  float        v8f;
typedef __attribute__((ext_vector_type(4)))  float        f32x4;
typedef __attribute__((ext_vector_type(4)))  unsigned int u32x4;
typedef int i32x4v __attribute__((vector_size(16)));      // matches builtin param

union FragAB { v16bf v; struct { u32x4 lo, hi; } q; };

// ---- CDNA5 async global->LDS path (guarded; plain copy fallback) ----------
#if defined(__has_builtin)
# if __has_builtin(__builtin_amdgcn_global_load_async_to_lds_b128)
#  define HAS_ASYNC_LDS 1
# endif
#endif
#ifndef HAS_ASYNC_LDS
# define HAS_ASYNC_LDS 0
#endif

#if HAS_ASYNC_LDS
#define AS_GLOBAL __attribute__((address_space(1)))
#define AS_LDS    __attribute__((address_space(3)))
# if __has_builtin(__builtin_amdgcn_s_wait_asynccnt)
#  define WAIT_ASYNC(n) __builtin_amdgcn_s_wait_asynccnt(n)
# else
#  define WAIT_ASYNC(n) asm volatile("s_wait_asynccnt %0" :: "i"(n) : "memory")
# endif
// copy 64 contiguous bytes global -> LDS via 4 async b128 ops (ASYNCcnt += 4)
__device__ __forceinline__ void async_copy64(const unsigned short* g, unsigned short* l) {
  AS_GLOBAL i32x4v* gp = (AS_GLOBAL i32x4v*)g;   // generic -> global AS, typed
  AS_LDS    i32x4v* lp = (AS_LDS    i32x4v*)l;   // generic -> LDS AS, typed
  __builtin_amdgcn_global_load_async_to_lds_b128(gp, lp,  0, 0);
  __builtin_amdgcn_global_load_async_to_lds_b128(gp, lp, 16, 0);
  __builtin_amdgcn_global_load_async_to_lds_b128(gp, lp, 32, 0);
  __builtin_amdgcn_global_load_async_to_lds_b128(gp, lp, 48, 0);
}
#else
#define WAIT_ASYNC(n) ((void)0)
#endif

__device__ __forceinline__ unsigned short f32_to_bf16(float f) {
  unsigned int u = __float_as_uint(f);
  u += 0x7FFFu + ((u >> 16) & 1u);
  return (unsigned short)(u >> 16);
}

// round-to-nearest pack of two f32 -> packed bf16x2 (cheap: ~2 VALU/elem)
__device__ __forceinline__ unsigned int pack_bf16x2(float a, float b) {
  unsigned int ua = __float_as_uint(a) + 0x8000u;
  unsigned int ub = __float_as_uint(b) + 0x8000u;
  return (ub & 0xFFFF0000u) | (ua >> 16);
}

// gfx1250 hardware tanh (TRANS op); v_nop covers the TRANS-result hazard
__device__ __forceinline__ float fast_tanh(float x) {
  float y;
  asm("v_tanh_f32 %0, %1\n\tv_nop" : "=v"(y) : "v"(x));
  return y;
}

// ---------------------------------------------------------------------------
// Kernel 1: qb[b,h] = dot(dec[b,:], W_w[h,:]) + V_b[h]; V_w -> bf16; zero ctx
// ---------------------------------------------------------------------------
__global__ __launch_bounds__(256)
void bahdanau_prep(const float* __restrict__ dec, const float* __restrict__ Ww,
                   const float* __restrict__ Vb,  const float* __restrict__ Vw,
                   float* __restrict__ qb, unsigned short* __restrict__ vwbf,
                   float* __restrict__ ctx)
{
  const int b = blockIdx.x, tid = threadIdx.x;
  if (tid < Hh) {
    const float* dr = dec + (size_t)b * Dd;
    const float* wr = Ww + (size_t)tid * Dd;
    float s = 0.f;
    for (int d = 0; d < Dd; ++d) s += dr[d] * wr[d];
    qb[b * Hh + tid] = s + Vb[tid];
  }
  const int base = b * 1024;                 // 64 blocks * 1024 = H*E
  for (int j = tid; j < 1024; j += 256)
    vwbf[base + j] = f32_to_bf16(Vw[base + j]);
  for (int j = tid; j < Ee; j += 256)        // zero context (atomics target)
    ctx[(size_t)b * Ee + j] = 0.f;
}

// ---------------------------------------------------------------------------
// Kernel 2: per-(b, 128-row T chunk) bf16 WMMA GEMM  k = enc @ V_w^T,
// register-double-buffered A, async-double-buffered B, fused tanh/score
// ---------------------------------------------------------------------------
__global__ __launch_bounds__(256)
void bahdanau_scores(const float* __restrict__ enc, const int* __restrict__ mask,
                     const float* __restrict__ qb,  const unsigned short* __restrict__ vwbf,
                     const float* __restrict__ ww,  const float* __restrict__ wbp,
                     float* __restrict__ scores)
{
  __shared__ unsigned short Abuf[128 * LDA];      // encoder tile (bf16)
  __shared__ unsigned short Bbuf[2][128 * LDA];   // V_w tile, double buffered
  __shared__ float qsh[Hh], wsh[Hh];

  const int tid    = threadIdx.x;
  const int lane   = tid & 31;
  const int wv     = tid >> 5;               // wave 0..7 -> 16-row strip
  const int col    = lane & 15;
  const bool lohalf = (lane < 16);
  const int b  = blockIdx.y;
  const int t0 = blockIdx.x * 128;

  if (tid < Hh) { qsh[tid] = qb[b * Hh + tid]; wsh[tid] = ww[tid]; }

  const int r  = tid >> 1;                   // LDS fill: row 0..127
  const int cb = (tid & 1) * 32;             // 32-col segment
  const float*          gA = enc  + ((size_t)b * Tt + t0 + r) * Ee + cb;
  const unsigned short* gB = vwbf + (size_t)r * Ee + cb;

  v8f zero = {0.f,0.f,0.f,0.f,0.f,0.f,0.f,0.f};
  v8f acc[8];
#pragma unroll
  for (int nt = 0; nt < 8; ++nt) acc[nt] = zero;

  // prologue: A(0) into regs; B(0) async into Bbuf[0]
  f32x4 aReg[2][8];
#pragma unroll
  for (int j = 0; j < 8; ++j)
    aReg[0][j] = __builtin_nontemporal_load((const f32x4*)gA + j);
#if HAS_ASYNC_LDS
  async_copy64(gB, &Bbuf[0][r * LDA + cb]);
#endif

  for (int kt = 0; kt < Ee / 64; ++kt) {
    const int pb = kt & 1;
    { // pack A(kt) f32 regs -> bf16 LDS (ds_store_b128 x4)
      u32x4* arow = (u32x4*)(Abuf + r * LDA + cb);
#pragma unroll
      for (int j = 0; j < 4; ++j) {
        f32x4 x = aReg[pb][2*j], y = aReg[pb][2*j + 1];
        u32x4 w;
        w[0] = pack_bf16x2(x[0], x[1]); w[1] = pack_bf16x2(x[2], x[3]);
        w[2] = pack_bf16x2(y[0], y[1]); w[3] = pack_bf16x2(y[2], y[3]);
        arow[j] = w;
      }
    }
#if !HAS_ASYNC_LDS
    { // fallback: plain B(kt) copy
      const u32x4* gb = (const u32x4*)(gB + kt * 64);
      u32x4* brow = (u32x4*)(&Bbuf[pb][r * LDA + cb]);
#pragma unroll
      for (int j = 0; j < 4; ++j) brow[j] = gb[j];
    }
#endif
    if (kt < Ee / 64 - 1) {
#if HAS_ASYNC_LDS
      async_copy64(gB + (kt + 1) * 64, &Bbuf[pb ^ 1][r * LDA + cb]);
#endif
#pragma unroll
      for (int j = 0; j < 8; ++j)   // prefetch A(kt+1) into regs
        aReg[pb ^ 1][j] = __builtin_nontemporal_load((const f32x4*)(gA + (kt + 1) * 64) + j);
      WAIT_ASYNC(4);                // B(kt) done (in-order), B(kt+1) in flight
    } else {
      WAIT_ASYNC(0);
    }
    __syncthreads();

#pragma unroll
    for (int ks = 0; ks < 2; ++ks) {
      const int ko = ks * 32;
      // A fragment (ISA bf16 A 16x32): lanes0-15 K{0..7,16..23}, lanes16-31 K{8..15,24..31}
      FragAB fa;
      const unsigned short* arow = Abuf + (wv * 16 + col) * LDA;
      fa.q.lo = *(const u32x4*)(arow + ko + (lohalf ? 0 : 8));
      fa.q.hi = *(const u32x4*)(arow + ko + (lohalf ? 16 : 24));
#pragma unroll
      for (int nt = 0; nt < 8; ++nt) {
        // B fragment 32x16: lanes0-15 K{0..7,16..23} of col N, lanes16-31 K{8..15,24..31}
        FragAB fb;
        const unsigned short* brow = &Bbuf[pb][(nt * 16 + col) * LDA + ko + (lohalf ? 0 : 8)];
        fb.q.lo = *(const u32x4*)(brow);
        fb.q.hi = *(const u32x4*)(brow + 16);
        acc[nt] = __builtin_amdgcn_wmma_f32_16x16x32_bf16(
            false, fa.v, false, fb.v, (short)0, acc[nt], false, false);
      }
    }
    __syncthreads();
  }

  // Epilogue: s[t] = sum_h tanh(k[t,h] + qb[h]) * ww[h] + wb ; mask -> -inf
  float rsum[8];
#pragma unroll
  for (int rr = 0; rr < 8; ++rr) rsum[rr] = 0.f;
#pragma unroll
  for (int nt = 0; nt < 8; ++nt) {
    const int h = nt * 16 + col;
    const float qv = qsh[h], wwv = wsh[h];
#pragma unroll
    for (int rr = 0; rr < 8; ++rr)
      rsum[rr] += fast_tanh(acc[nt][rr] + qv) * wwv;
  }
#pragma unroll
  for (int rr = 0; rr < 8; ++rr) {
#pragma unroll
    for (int m = 8; m >= 1; m >>= 1)               // reduce within 16-lane halves
      rsum[rr] += __shfl_xor(rsum[rr], m, 32);
  }
  if (col == 0) {                                  // lanes 0 and 16: rows rr / rr+8
    const float wb = wbp[0];
#pragma unroll
    for (int rr = 0; rr < 8; ++rr) {
      const int t = t0 + wv * 16 + rr + (lohalf ? 0 : 8);
      float sc = rsum[rr] + wb;
      if (mask[(size_t)b * Tt + t] != 0) sc = -__builtin_inff();
      scores[(size_t)b * Tt + t] = sc;
    }
  }
}

// ---------------------------------------------------------------------------
// Kernel 3: softmax over T per batch row
// ---------------------------------------------------------------------------
__global__ __launch_bounds__(256)
void bahdanau_softmax(const float* __restrict__ scores, float* __restrict__ attn)
{
  __shared__ float red[256];
  const int b = blockIdx.x, tid = threadIdx.x;
  float v[8];
#pragma unroll
  for (int i = 0; i < 8; ++i) v[i] = scores[(size_t)b * Tt + i * 256 + tid];
  float m = v[0];
#pragma unroll
  for (int i = 1; i < 8; ++i) m = fmaxf(m, v[i]);
  red[tid] = m; __syncthreads();
  for (int s = 128; s > 0; s >>= 1) {
    if (tid < s) red[tid] = fmaxf(red[tid], red[tid + s]);
    __syncthreads();
  }
  const float gm = red[0];
  __syncthreads();
  float e[8]; float sum = 0.f;
#pragma unroll
  for (int i = 0; i < 8; ++i) { e[i] = expf(v[i] - gm); sum += e[i]; }
  red[tid] = sum; __syncthreads();
  for (int s = 128; s > 0; s >>= 1) {
    if (tid < s) red[tid] += red[tid + s];
    __syncthreads();
  }
  const float inv = 1.f / red[0];
#pragma unroll
  for (int i = 0; i < 8; ++i)
    attn[(size_t)b * Tt + i * 256 + tid] = e[i] * inv;
}

// ---------------------------------------------------------------------------
// Kernel 4: context[b,e] = sum_t enc[b,t,e] * attn[b,t] (partials via atomics)
// ---------------------------------------------------------------------------
__global__ __launch_bounds__(256)
void bahdanau_context(const float* __restrict__ enc, const float* __restrict__ attn,
                      float* __restrict__ ctx)
{
  __shared__ float ash[256];
  const int b  = blockIdx.y;
  const int es = blockIdx.x & 1;            // which 256-wide E half
  const int ts = blockIdx.x >> 1;           // which 256-long T slice (0..7)
  const int e  = es * 256 + threadIdx.x;
  const int tb = ts * 256;
  ash[threadIdx.x] = attn[(size_t)b * Tt + tb + threadIdx.x];
  __syncthreads();
  float sum = 0.f;
  for (int i = 0; i < 256; ++i)
    sum += __builtin_nontemporal_load(enc + ((size_t)b * Tt + tb + i) * Ee + e) * ash[i];
  atomicAdd(&ctx[(size_t)b * Ee + e], sum);
}

// ---------------------------------------------------------------------------
extern "C" void kernel_launch(void* const* d_in, const int* in_sizes, int n_in,
                              void* d_out, int out_size, void* d_ws, size_t ws_size,
                              hipStream_t stream)
{
  (void)in_sizes; (void)n_in; (void)out_size; (void)ws_size;
  const float* enc  = (const float*)d_in[0];  // [B,T,E]
  const float* dec  = (const float*)d_in[1];  // [B,D]
  const int*   mask = (const int*)d_in[2];    // [B,T]
  const float* Vw   = (const float*)d_in[3];  // [H,E]
  const float* Vb   = (const float*)d_in[4];  // [H]
  const float* Ww   = (const float*)d_in[5];  // [H,D]
  const float* ww   = (const float*)d_in[6];  // [1,H]
  const float* wb   = (const float*)d_in[7];  // [1]

  float* out  = (float*)d_out;
  float* ctx  = out;                 // [B,E]   (tuple output 0)
  float* attn = out + Bb * Ee;       // [B,T]   (tuple output 1)

  float*          qb     = (float*)d_ws;                          // B*H
  float*          scores = qb + Bb * Hh;                          // B*T
  unsigned short* vwbf   = (unsigned short*)(scores + Bb * Tt);   // H*E bf16

  bahdanau_prep   <<<Bb, 256, 0, stream>>>(dec, Ww, Vb, Vw, qb, vwbf, ctx);
  bahdanau_scores <<<dim3(Tt / 128, Bb), 256, 0, stream>>>(enc, mask, qb, vwbf, ww, wb, scores);
  bahdanau_softmax<<<Bb, 256, 0, stream>>>(scores, attn);
  bahdanau_context<<<dim3(16, Bb), 256, 0, stream>>>(enc, attn, ctx);
}
